// DecoderBlock_88656714925041
// MI455X (gfx1250) — matrix operand order repaired and checked by reading, not compiled, not run on hardware
//
#include <hip/hip_runtime.h>
#include <hip/hip_bf16.h>

typedef _Float16 h16_t;
typedef __attribute__((ext_vector_type(16))) _Float16 v16h;
typedef __attribute__((ext_vector_type(8)))  float    v8f;

#define D_MODEL 1024
#define NHEADS  16
#define HDIM    64

static __device__ __forceinline__ v8f wmma32(v16h a, v16h b, v8f c) {
  // D = A(16x32 f16) * B(32x16 f16) + C(16x16 f32)
  return __builtin_amdgcn_wmma_f32_16x16x32_f16(false, a, false, b, (short)0, c, false, false);
}

// Async global->LDS 16B copy (CDNA5 TDM-class path, tracked by ASYNCcnt).
// LDS operand is the wave-relative LDS byte offset = low 32 bits of the flat
// shared-aperture address (see ISA 10.2).
static __device__ __forceinline__ void async_cp16(const void* gptr, void* lptr) {
  unsigned lds = (unsigned)(uintptr_t)lptr;
  asm volatile("global_load_async_to_lds_b128 %0, %1, off"
               :: "v"(lds), "v"(gptr) : "memory");
}

// ---------------------------------------------------------------------------
// fp32 -> f16 elementwise convert
// ---------------------------------------------------------------------------
__global__ void cvt_f32_f16(const float* __restrict__ src, h16_t* __restrict__ dst, int n) {
  int i = blockIdx.x * blockDim.x + threadIdx.x;
  if (i < n) dst[i] = (h16_t)src[i];
}

// ---------------------------------------------------------------------------
// GEMM: C[rows, OUT] = A[rows, K](f16) @ W[OUT, K](f16)^T + bias
// Block tile 64x256, 8 waves as 2x4 grid of 32x64 wave tiles, K step 64.
// Double-buffered async global->LDS copies overlap the WMMA work.
// rows % 64 == 0, OUT % 256 == 0, K % 64 == 0 (true for all uses here).
// ---------------------------------------------------------------------------
__global__ __launch_bounds__(256)
void gemm_f16(const h16_t* __restrict__ A, const h16_t* __restrict__ W,
              const float* __restrict__ bias, h16_t* __restrict__ out16,
              float* __restrict__ out32, int K, int OUT, int relu)
{
  // stride 72 f16 (144 B = 9x16 B): keeps b128 gathers aligned, 2-way max
  __shared__ __align__(16) h16_t sA[2][64][72];
  __shared__ __align__(16) h16_t sB[2][256][72];

  const int tid  = threadIdx.x;
  const int lane = tid & 31;
  const int wave = tid >> 5;
  const int wr   = wave >> 2;          // 0..1 : 32-row quadrant
  const int wc   = wave & 3;           // 0..3 : 64-col quadrant
  const int row0 = blockIdx.y * 64;
  const int col0 = blockIdx.x * 256;
  const int l15  = lane & 15;
  const bool lo  = lane < 16;

  v8f acc[2][4];
#pragma unroll
  for (int s = 0; s < 2; ++s)
#pragma unroll
    for (int t = 0; t < 4; ++t)
      acc[s][t] = (v8f){0.f,0.f,0.f,0.f,0.f,0.f,0.f,0.f};

  // Per-tile async copy: A 64x64 f16 = 512 16B chunks (2/thread),
  //                      W 256x64 f16 = 2048 chunks (8/thread) -> 10 ops/wave.
  auto issue = [&](int kt, int buf) {
    const int kbase = kt * 64;
#pragma unroll
    for (int i = 0; i < 2; ++i) {
      const int chunk = tid + i * 256;
      const int r = chunk >> 3, kc = (chunk & 7) * 8;
      async_cp16(A + (size_t)(row0 + r) * K + kbase + kc, &sA[buf][r][kc]);
    }
#pragma unroll
    for (int i = 0; i < 8; ++i) {
      const int chunk = tid + i * 256;
      const int c = chunk >> 3, kc = (chunk & 7) * 8;
      async_cp16(W + (size_t)(col0 + c) * K + kbase + kc, &sB[buf][c][kc]);
    }
  };

  const int nk = K >> 6;
  issue(0, 0);

  for (int kt = 0; kt < nk; ++kt) {
    const int cur = kt & 1;
    if (kt + 1 < nk) {
      issue(kt + 1, cur ^ 1);
      asm volatile("s_wait_asynccnt 10" ::: "memory");   // batch kt landed
    } else {
      asm volatile("s_wait_asynccnt 0" ::: "memory");
    }
    __syncthreads();

#pragma unroll
    for (int kc = 0; kc < 2; ++kc) {                      // k sub-chunks of 32
      v16h af[2], bf[4];
#pragma unroll
      for (int s = 0; s < 2; ++s) {
        const h16_t* p = &sA[cur][wr * 32 + s * 16 + l15][kc * 32 + (lo ? 0 : 16)];
#pragma unroll
        for (int i = 0; i < 8; ++i) { af[s][i] = p[i]; af[s][8 + i] = p[8 + i]; }
      }
#pragma unroll
      for (int t = 0; t < 4; ++t) {
        const h16_t* p = &sB[cur][wc * 64 + t * 16 + l15][kc * 32 + (lo ? 0 : 16)];
#pragma unroll
        for (int i = 0; i < 16; ++i) bf[t][i] = p[i];
      }
#pragma unroll
      for (int s = 0; s < 2; ++s)
#pragma unroll
        for (int t = 0; t < 4; ++t)
          acc[s][t] = wmma32(af[s], bf[t], acc[s][t]);
    }
    __syncthreads();
  }

  // Epilogue: C layout -> VGPR r holds (row = r or r+8, col = lane&15)
  const int rhalf = lo ? 0 : 8;
#pragma unroll
  for (int t = 0; t < 4; ++t) {
    const int col = col0 + wc * 64 + t * 16 + l15;
    const float bv = bias ? bias[col] : 0.f;
#pragma unroll
    for (int s = 0; s < 2; ++s) {
#pragma unroll
      for (int r = 0; r < 8; ++r) {
        const int row = row0 + wr * 32 + s * 16 + r + rhalf;
        float v = acc[s][t][r] + bv;
        if (relu) v = v > 0.f ? v : 0.f;
        if (out32) out32[(size_t)row * OUT + col] = v;
        if (out16) out16[(size_t)row * OUT + col] = (h16_t)v;
      }
    }
  }
}

// ---------------------------------------------------------------------------
// Flash attention: 1 wave = 16 query rows, block = 64 rows for one (b, head).
// Per 32-key block: S = Q K^T (4 WMMA), online softmax, O += P V (4 WMMA).
// ---------------------------------------------------------------------------
__global__ __launch_bounds__(128)
void flash_attn(const h16_t* __restrict__ Q, const h16_t* __restrict__ Kk,
                const h16_t* __restrict__ V, h16_t* __restrict__ O,
                int qlen, int kvlen, int causal)
{
  __shared__ __align__(16) h16_t pbuf[4][16][40];  // per-wave P reshape staging
  const int lane = threadIdx.x & 31;
  const int w    = threadIdx.x >> 5;
  const int h    = blockIdx.y;
  const int b    = blockIdx.z;
  const int q0   = blockIdx.x * 64 + w * 16;
  const int l15  = lane & 15;
  const bool lo  = lane < 16;
  const size_t kvb = (size_t)b * kvlen;

  // Q A-fragments (K-dim = 64 -> two 16x32 fragments), held for whole kernel
  v16h aq0, aq1;
  {
    const h16_t* qp = Q + (size_t)(b * qlen + q0 + l15) * D_MODEL + h * HDIM;
    const int ko = lo ? 0 : 16;
#pragma unroll
    for (int i = 0; i < 8; ++i) {
      aq0[i]     = qp[ko + i];       aq0[8 + i] = qp[ko + 8 + i];
      aq1[i]     = qp[32 + ko + i];  aq1[8 + i] = qp[32 + ko + 8 + i];
    }
  }

  float m[8], l[8];
  v8f o_acc[4];
#pragma unroll
  for (int r = 0; r < 8; ++r) { m[r] = -1e38f; l[r] = 0.f; }
#pragma unroll
  for (int ct = 0; ct < 4; ++ct) o_acc[ct] = (v8f){0.f,0.f,0.f,0.f,0.f,0.f,0.f,0.f};

  const int lastk = causal ? (q0 + 16) : kvlen;
  const int nkb   = (lastk + 31) >> 5;

  for (int kb = 0; kb < nkb; ++kb) {
    const int kb0 = kb * 32;
    v8f s[2];
#pragma unroll
    for (int hh = 0; hh < 2; ++hh) {
      s[hh] = (v8f){0.f,0.f,0.f,0.f,0.f,0.f,0.f,0.f};
      const int key = kb0 + hh * 16 + l15;           // B-frag: lane = key col
      const h16_t* kp = Kk + (kvb + key) * D_MODEL + h * HDIM + (lo ? 0 : 16);
      v16h bk0, bk1;
#pragma unroll
      for (int i = 0; i < 16; ++i) { bk0[i] = kp[i]; bk1[i] = kp[32 + i]; }
      s[hh] = wmma32(aq0, bk0, s[hh]);
      s[hh] = wmma32(aq1, bk1, s[hh]);
    }
    // scale + causal mask (analytic tril == reference mask)
#pragma unroll
    for (int hh = 0; hh < 2; ++hh) {
      const int key = kb0 + hh * 16 + l15;
#pragma unroll
      for (int r = 0; r < 8; ++r) {
        float v = s[hh][r] * 0.125f;                 // 1/sqrt(64)
        if (causal && key > (q0 + r + (lo ? 0 : 8))) v = -1e30f;
        s[hh][r] = v;
      }
    }
    // online softmax; xor 1/2/4/8 stays inside the 16-lane half owning the row
    float p0[8], p1[8];
#pragma unroll
    for (int r = 0; r < 8; ++r) {
      float v = fmaxf(s[0][r], s[1][r]);
      v = fmaxf(v, __shfl_xor(v, 1)); v = fmaxf(v, __shfl_xor(v, 2));
      v = fmaxf(v, __shfl_xor(v, 4)); v = fmaxf(v, __shfl_xor(v, 8));
      const float mn = fmaxf(m[r], v);
      const float sc = __expf(m[r] - mn);
      m[r] = mn; l[r] *= sc;
#pragma unroll
      for (int ct = 0; ct < 4; ++ct) o_acc[ct][r] *= sc;
      p0[r] = __expf(s[0][r] - mn);
      p1[r] = __expf(s[1][r] - mn);
      float t = p0[r] + p1[r];
      t += __shfl_xor(t, 1); t += __shfl_xor(t, 2);
      t += __shfl_xor(t, 4); t += __shfl_xor(t, 8);
      l[r] += t;
    }
    // P (C layout) -> LDS -> A-fragment layout
#pragma unroll
    for (int r = 0; r < 8; ++r) {
      const int rl = r + (lo ? 0 : 8);
      pbuf[w][rl][l15]      = (h16_t)p0[r];
      pbuf[w][rl][l15 + 16] = (h16_t)p1[r];
    }
    asm volatile("s_wait_dscnt 0" ::: "memory");     // CDNA5 split DS counter wait
    v16h ap;
    {
      const h16_t* pp = &pbuf[w][l15][lo ? 0 : 16];
#pragma unroll
      for (int i = 0; i < 8; ++i) { ap[i] = pp[i]; ap[8 + i] = pp[8 + i]; }
    }
    // O += P(16x32) * V(32x64) : 4 column tiles
#pragma unroll
    for (int ct = 0; ct < 4; ++ct) {
      v16h bv;
      const h16_t* vp = V + (kvb + kb0 + (lo ? 0 : 16)) * D_MODEL + h * HDIM + ct * 16 + l15;
#pragma unroll
      for (int kk = 0; kk < 16; ++kk) bv[kk] = vp[(size_t)kk * D_MODEL];
      o_acc[ct] = wmma32(ap, bv, o_acc[ct]);
    }
  }

#pragma unroll
  for (int r = 0; r < 8; ++r) {
    const float inv = 1.f / l[r];
    const int qrow  = q0 + r + (lo ? 0 : 8);
#pragma unroll
    for (int ct = 0; ct < 4; ++ct)
      O[(size_t)(b * qlen + qrow) * D_MODEL + h * HDIM + ct * 16 + l15] =
          (h16_t)(o_acc[ct][r] * inv);
  }
}

// ---------------------------------------------------------------------------
// out = LayerNorm(a + bres) * g + beta ; also emits f16 mirror for next GEMM
// ---------------------------------------------------------------------------
__global__ __launch_bounds__(256)
void add_ln(const float* __restrict__ a, const float* __restrict__ bres,
            const float* __restrict__ g, const float* __restrict__ beta,
            float* __restrict__ out32, h16_t* __restrict__ out16)
{
  __shared__ float red[2][8];
  const int row = blockIdx.x;
  const int tid = threadIdx.x;
  float v[4], s = 0.f, s2 = 0.f;
#pragma unroll
  for (int i = 0; i < 4; ++i) {
    const int c = tid + i * 256;
    v[i] = a[(size_t)row * D_MODEL + c] + bres[(size_t)row * D_MODEL + c];
    s += v[i]; s2 += v[i] * v[i];
  }
#pragma unroll
  for (int off = 16; off > 0; off >>= 1) { s += __shfl_xor(s, off); s2 += __shfl_xor(s2, off); }
  if ((tid & 31) == 0) { red[0][tid >> 5] = s; red[1][tid >> 5] = s2; }
  __syncthreads();
  float ts = 0.f, ts2 = 0.f;
#pragma unroll
  for (int i = 0; i < 8; ++i) { ts += red[0][i]; ts2 += red[1][i]; }
  const float mu  = ts * (1.f / D_MODEL);
  const float var = ts2 * (1.f / D_MODEL) - mu * mu;
  const float rs  = rsqrtf(var + 1e-5f);
#pragma unroll
  for (int i = 0; i < 4; ++i) {
    const int c = tid + i * 256;
    const float o = (v[i] - mu) * rs * g[c] + beta[c];
    out32[(size_t)row * D_MODEL + c] = o;
    if (out16) out16[(size_t)row * D_MODEL + c] = (h16_t)o;
  }
}

// ---------------------------------------------------------------------------
extern "C" void kernel_launch(void* const* d_in, const int* in_sizes, int n_in,
                              void* d_out, int out_size, void* d_ws, size_t ws_size,
                              hipStream_t stream)
{
  (void)in_sizes; (void)n_in; (void)out_size; (void)ws_size;
  const int B = 2, N = 2048, M = 1024, D = D_MODEL, FF = 4096;
  const int BN = B * N;   // 4096 rows
  const int BM = B * M;   // 2048 rows

  const float* x      = (const float*)d_in[0];
  const float* memory = (const float*)d_in[1];
  // d_in[2] = tril mask, implemented analytically in flash_attn
  const float* sa_wq = (const float*)d_in[3];  const float* sa_bq = (const float*)d_in[4];
  const float* sa_wk = (const float*)d_in[5];  const float* sa_bk = (const float*)d_in[6];
  const float* sa_wv = (const float*)d_in[7];  const float* sa_bv = (const float*)d_in[8];
  const float* sa_wo = (const float*)d_in[9];  const float* sa_bo = (const float*)d_in[10];
  const float* ca_wq = (const float*)d_in[11]; const float* ca_bq = (const float*)d_in[12];
  const float* ca_wk = (const float*)d_in[13]; const float* ca_bk = (const float*)d_in[14];
  const float* ca_wv = (const float*)d_in[15]; const float* ca_bv = (const float*)d_in[16];
  const float* ca_wo = (const float*)d_in[17]; const float* ca_bo = (const float*)d_in[18];
  const float* ff_w1 = (const float*)d_in[19]; const float* ff_b1 = (const float*)d_in[20];
  const float* ff_w2 = (const float*)d_in[21]; const float* ff_b2 = (const float*)d_in[22];
  const float* ln1_g = (const float*)d_in[23]; const float* ln1_b = (const float*)d_in[24];
  const float* ln2_g = (const float*)d_in[25]; const float* ln2_b = (const float*)d_in[26];
  const float* ln3_g = (const float*)d_in[27]; const float* ln3_b = (const float*)d_in[28];

  char* ws = (char*)d_ws;
  size_t off = 0;
  auto alloc = [&](size_t bytes) -> void* {
    void* p = ws + off; off += (bytes + 255) & ~(size_t)255; return p;
  };
  h16_t* x16   = (h16_t*)alloc((size_t)BN * D * 2);
  h16_t* mem16 = (h16_t*)alloc((size_t)BM * D * 2);
  h16_t* q16   = (h16_t*)alloc((size_t)BN * D * 2);
  h16_t* k16   = (h16_t*)alloc((size_t)BN * D * 2);
  h16_t* v16   = (h16_t*)alloc((size_t)BN * D * 2);
  h16_t* att16 = (h16_t*)alloc((size_t)BN * D * 2);
  float* tmp32 = (float*)alloc((size_t)BN * D * 4);
  float* x1_32 = (float*)alloc((size_t)BN * D * 4);
  h16_t* x1_16 = (h16_t*)alloc((size_t)BN * D * 2);
  float* x2_32 = (float*)alloc((size_t)BN * D * 4);
  h16_t* x2_16 = (h16_t*)alloc((size_t)BN * D * 2);
  h16_t* hbuf  = (h16_t*)alloc((size_t)BN * FF * 2);
  // f16 weight mirrors (all weights fit in the 192MB L2 either way)
  h16_t* hsa_wq = (h16_t*)alloc((size_t)D * D * 2);
  h16_t* hsa_wk = (h16_t*)alloc((size_t)D * D * 2);
  h16_t* hsa_wv = (h16_t*)alloc((size_t)D * D * 2);
  h16_t* hsa_wo = (h16_t*)alloc((size_t)D * D * 2);
  h16_t* hca_wq = (h16_t*)alloc((size_t)D * D * 2);
  h16_t* hca_wk = (h16_t*)alloc((size_t)D * D * 2);
  h16_t* hca_wv = (h16_t*)alloc((size_t)D * D * 2);
  h16_t* hca_wo = (h16_t*)alloc((size_t)D * D * 2);
  h16_t* hff_w1 = (h16_t*)alloc((size_t)FF * D * 2);
  h16_t* hff_w2 = (h16_t*)alloc((size_t)D * FF * 2);

  const int DD = D * D, DF = D * FF;
  const int cvb = 256;
  cvt_f32_f16<<<(BN * D + cvb - 1) / cvb, cvb, 0, stream>>>(x,      x16,   BN * D);
  cvt_f32_f16<<<(BM * D + cvb - 1) / cvb, cvb, 0, stream>>>(memory, mem16, BM * D);
  cvt_f32_f16<<<(DD + cvb - 1) / cvb, cvb, 0, stream>>>(sa_wq, hsa_wq, DD);
  cvt_f32_f16<<<(DD + cvb - 1) / cvb, cvb, 0, stream>>>(sa_wk, hsa_wk, DD);
  cvt_f32_f16<<<(DD + cvb - 1) / cvb, cvb, 0, stream>>>(sa_wv, hsa_wv, DD);
  cvt_f32_f16<<<(DD + cvb - 1) / cvb, cvb, 0, stream>>>(sa_wo, hsa_wo, DD);
  cvt_f32_f16<<<(DD + cvb - 1) / cvb, cvb, 0, stream>>>(ca_wq, hca_wq, DD);
  cvt_f32_f16<<<(DD + cvb - 1) / cvb, cvb, 0, stream>>>(ca_wk, hca_wk, DD);
  cvt_f32_f16<<<(DD + cvb - 1) / cvb, cvb, 0, stream>>>(ca_wv, hca_wv, DD);
  cvt_f32_f16<<<(DD + cvb - 1) / cvb, cvb, 0, stream>>>(ca_wo, hca_wo, DD);
  cvt_f32_f16<<<(DF + cvb - 1) / cvb, cvb, 0, stream>>>(ff_w1, hff_w1, DF);
  cvt_f32_f16<<<(DF + cvb - 1) / cvb, cvb, 0, stream>>>(ff_w2, hff_w2, DF);

  const dim3 gN (D  / 256, BN / 64);   // rows=4096, OUT=1024
  const dim3 gM (D  / 256, BM / 64);   // rows=2048, OUT=1024
  const dim3 gF1(FF / 256, BN / 64);   // rows=4096, OUT=4096
  const dim3 gAtt(N / 64, NHEADS, B);

  // ---- self attention ----
  gemm_f16<<<gN, 256, 0, stream>>>(x16, hsa_wq, sa_bq, q16, nullptr, D, D, 0);
  gemm_f16<<<gN, 256, 0, stream>>>(x16, hsa_wk, sa_bk, k16, nullptr, D, D, 0);
  gemm_f16<<<gN, 256, 0, stream>>>(x16, hsa_wv, sa_bv, v16, nullptr, D, D, 0);
  flash_attn<<<gAtt, 128, 0, stream>>>(q16, k16, v16, att16, N, N, 1);
  gemm_f16<<<gN, 256, 0, stream>>>(att16, hsa_wo, sa_bo, nullptr, tmp32, D, D, 0);
  add_ln<<<BN, 256, 0, stream>>>(x, tmp32, ln1_g, ln1_b, x1_32, x1_16);

  // ---- cross attention ----
  gemm_f16<<<gN, 256, 0, stream>>>(x1_16, hca_wq, ca_bq, q16, nullptr, D, D, 0);
  gemm_f16<<<gM, 256, 0, stream>>>(mem16, hca_wk, ca_bk, k16, nullptr, D, D, 0);
  gemm_f16<<<gM, 256, 0, stream>>>(mem16, hca_wv, ca_bv, v16, nullptr, D, D, 0);
  flash_attn<<<gAtt, 128, 0, stream>>>(q16, k16, v16, att16, N, M, 0);
  gemm_f16<<<gN, 256, 0, stream>>>(att16, hca_wo, ca_bo, nullptr, tmp32, D, D, 0);
  add_ln<<<BN, 256, 0, stream>>>(x1_32, tmp32, ln2_g, ln2_b, x2_32, x2_16);

  // ---- feed-forward ----
  gemm_f16<<<gF1, 256, 0, stream>>>(x2_16, hff_w1, ff_b1, hbuf, nullptr, D, FF, 1);
  gemm_f16<<<gN,  256, 0, stream>>>(hbuf,  hff_w2, ff_b2, nullptr, tmp32, FF, D, 0);
  add_ln<<<BN, 256, 0, stream>>>(x2_32, tmp32, ln3_g, ln3_b, (float*)d_out, nullptr);
}
